// PaperDONNSentimentClassifier_76862734729918
// MI455X (gfx1250) — matrix operand order repaired and checked by reading, not compiled, not run on hardware
//
#include <hip/hip_runtime.h>
#include <hip/hip_bf16.h>

typedef __attribute__((ext_vector_type(16))) _Float16 v16h;
typedef __attribute__((ext_vector_type(8)))  _Float16 v8h;
typedef __attribute__((ext_vector_type(8)))  float    v8f;

#define T_LEN 1024
#define U_DIM 100
#define E_DIM 100
#define P_DIM 20
#define C_DIM 2
#define K_PAD 224   // 7 * 32  (2U=200 padded)
#define N_PAD 112   // 7 * 16  (U=100 padded)
#define NOSC  7     // oscillators per thread (16x16 thread map covers 16x112 >= 16x100)
#define HDT   0.001f
#define HSCALE 0.2f
#define HMU   1.0f

__global__ __launch_bounds__(256)
void donn_hopf_kernel(const int*   __restrict__ x,
                      const float* __restrict__ emb,
                      const float* __restrict__ omega1,
                      const float* __restrict__ omega2,
                      const float* __restrict__ w1, const float* __restrict__ b1,
                      const float* __restrict__ w2, const float* __restrict__ b2,
                      const float* __restrict__ wp, const float* __restrict__ bp,
                      const float* __restrict__ wh, const float* __restrict__ bh,
                      float* __restrict__ out)
{
    __shared__ _Float16 Abuf[16][K_PAD];   // [z_r | z_i] tile, f16, zero-padded K
    __shared__ float    Hbuf[16][N_PAD];   // h1 (and later h2), f32
    __shared__ float    H3[16][P_DIM];

    const int tid  = threadIdx.x;
    const int lane = tid & 31;
    const int w    = tid >> 5;          // wave id 0..7 (0..6 do WMMA)
    const int b0   = blockIdx.x * 16;   // batch tile base

    // elementwise (b,u) mapping: 16 rows x 16 u-groups
    const int bl = tid & 15;
    const int ub = tid >> 4;

    // zero-init A tile (padding columns stay zero forever)
    for (int i = tid; i < 16 * K_PAD; i += 256)
        ((_Float16*)Abuf)[i] = (_Float16)0.f;

    // WMMA column mapping for this lane
    const int n  = w * 16 + (lane & 15);   // output column (wave 7 -> >=112, masked off)
    const int hh = lane >> 4;              // lane half
    const float bias1 = (n < U_DIM) ? b1[n] : 0.f;
    const float bias2 = (n < U_DIM) ? b2[n] : 0.f;

    // stationary B-fragments for w1 [200,100] -> padded [224,112], f16, live in VGPRs
    v16h bw[7];
    #pragma unroll
    for (int kt = 0; kt < 7; ++kt) {
        v16h f;
        #pragma unroll
        for (int i = 0; i < 16; ++i) {
            int k = kt * 32 + hh * 16 + i;               // contiguous k = 16*hh + i, same order as the A loads
            float v = (k < 2 * U_DIM && n < U_DIM) ? w1[k * U_DIM + n] : 0.f;
            f[i] = (_Float16)v;
        }
        bw[kt] = f;
    }

    // oscillator state in registers
    float z1x[NOSC], z1y[NOSC], z2x[NOSC], z2y[NOSC];
    float om1[NOSC], om2[NOSC];
    int   uu[NOSC];
    #pragma unroll
    for (int k = 0; k < NOSC; ++k) {
        int u = ub + 16 * k;
        uu[k]  = u;
        z1x[k] = z1y[k] = z2x[k] = z2y[k] = 0.f;
        om1[k] = (u < U_DIM) ? omega1[u] : 0.f;
        om2[k] = (u < U_DIM) ? omega2[u] : 0.f;
    }

    __syncthreads();

    const long xrow = (long)(b0 + bl) * T_LEN;

    for (int t = 0; t < T_LEN; ++t) {
        // prefetch next timestep's embedding rows into cache (global_prefetch_b8)
        if (tid < 16 && t + 1 < T_LEN) {
            int xin = x[(long)(b0 + tid) * T_LEN + t + 1];
            __builtin_prefetch(emb + (long)xin * E_DIM, 0, 3);
        }

        // ---- layer-1 Hopf: 5 Euler steps driven by emb gather ----
        {
            const int xi = x[xrow + t];
            const float* erow = emb + (long)xi * E_DIM;
            #pragma unroll
            for (int k = 0; k < NOSC; ++k) {
                if (uu[k] < U_DIM) {
                    float d  = HSCALE * erow[uu[k]];
                    float xx = z1x[k], yy = z1y[k], om = om1[k];
                    #pragma unroll
                    for (int s = 0; s < 5; ++s) {
                        float r2 = xx * xx + yy * yy;
                        float nx = xx + HDT * ((HMU - r2) * xx - om * yy + d);
                        float ny = yy + HDT * ((HMU - r2) * yy + om * xx);
                        xx = nx; yy = ny;
                    }
                    z1x[k] = xx; z1y[k] = yy;
                    Abuf[bl][uu[k]]         = (_Float16)xx;
                    Abuf[bl][U_DIM + uu[k]] = (_Float16)yy;
                }
            }
        }
        __syncthreads();

        // ---- h1 = relu([z1r|z1i] @ w1 + b1) : 7 chained WMMAs per wave ----
        if (w < 7) {
            const int m = lane & 15;
            // issue ALL 14 independent A-fragment loads first: one LDS
            // round-trip instead of seven on the serial accumulator chain
            v8h fr[14];
            #pragma unroll
            for (int kt = 0; kt < 7; ++kt) {
                fr[2 * kt]     = *(const v8h*)&Abuf[m][kt * 32 + hh * 16];
                fr[2 * kt + 1] = *(const v8h*)&Abuf[m][kt * 32 + hh * 16 + 8];
            }
            // bias rides in C for free: D = A*B + C with C = bias broadcast
            v8f acc;
            #pragma unroll
            for (int v = 0; v < 8; ++v) acc[v] = bias1;
            #pragma unroll
            for (int kt = 0; kt < 7; ++kt) {
                v16h a = __builtin_shufflevector(fr[2 * kt], fr[2 * kt + 1],
                          0,1,2,3,4,5,6,7,8,9,10,11,12,13,14,15);
                acc = __builtin_amdgcn_wmma_f32_16x16x32_f16(
                          false, a, false, bw[kt], (short)0, acc, false, false);
            }
            #pragma unroll
            for (int v = 0; v < 8; ++v) {
                float val = acc[v];
                Hbuf[v + 8 * hh][n] = val > 0.f ? val : 0.f;
            }
        }
        __syncthreads();

        // ---- layer-2 Hopf: 5 Euler steps driven by h1 ----
        #pragma unroll
        for (int k = 0; k < NOSC; ++k) {
            if (uu[k] < U_DIM) {
                float d  = HSCALE * Hbuf[bl][uu[k]];
                float xx = z2x[k], yy = z2y[k], om = om2[k];
                #pragma unroll
                for (int s = 0; s < 5; ++s) {
                    float r2 = xx * xx + yy * yy;
                    float nx = xx + HDT * ((HMU - r2) * xx - om * yy + d);
                    float ny = yy + HDT * ((HMU - r2) * yy + om * xx);
                    xx = nx; yy = ny;
                }
                z2x[k] = xx; z2y[k] = yy;
            }
        }
        // next iter's Abuf writes are safe: last Abuf reads were before the 2nd barrier
    }

    // ---- final-timestep head: h2 = relu([z2r|z2i] @ w2 + b2) ----
    #pragma unroll
    for (int k = 0; k < NOSC; ++k) {
        if (uu[k] < U_DIM) {
            Abuf[bl][uu[k]]         = (_Float16)z2x[k];
            Abuf[bl][U_DIM + uu[k]] = (_Float16)z2y[k];
        }
    }
    __syncthreads();

    if (w < 7) {
        const int m = lane & 15;
        v8h fr[14];
        #pragma unroll
        for (int kt = 0; kt < 7; ++kt) {
            fr[2 * kt]     = *(const v8h*)&Abuf[m][kt * 32 + hh * 16];
            fr[2 * kt + 1] = *(const v8h*)&Abuf[m][kt * 32 + hh * 16 + 8];
        }
        v8f acc;
        #pragma unroll
        for (int v = 0; v < 8; ++v) acc[v] = bias2;
        #pragma unroll
        for (int kt = 0; kt < 7; ++kt) {
            v16h f;
            #pragma unroll
            for (int i = 0; i < 16; ++i) {
                int k = kt * 32 + hh * 16 + i;
                float v = (k < 2 * U_DIM && n < U_DIM) ? w2[k * U_DIM + n] : 0.f;
                f[i] = (_Float16)v;
            }
            v16h a = __builtin_shufflevector(fr[2 * kt], fr[2 * kt + 1],
                      0,1,2,3,4,5,6,7,8,9,10,11,12,13,14,15);
            acc = __builtin_amdgcn_wmma_f32_16x16x32_f16(
                      false, a, false, f, (short)0, acc, false, false);
        }
        #pragma unroll
        for (int v = 0; v < 8; ++v) {
            float val = acc[v];
            Hbuf[v + 8 * hh][n] = val > 0.f ? val : 0.f;
        }
    }
    __syncthreads();

    // ---- h3 = tanh(h2 @ wp + bp) : [16,100]x[100,20] ----
    for (int idx = tid; idx < 16 * P_DIM; idx += 256) {
        int b = idx / P_DIM, p = idx % P_DIM;
        float s = bp[p];
        #pragma unroll 4
        for (int j = 0; j < U_DIM; ++j)
            s += Hbuf[b][j] * wp[j * P_DIM + p];
        H3[b][p] = tanhf(s);
    }
    __syncthreads();

    // ---- out = h3 @ wh + bh : [16,20]x[20,2] ----
    if (tid < 16 * C_DIM) {
        int b = tid >> 1, c = tid & 1;
        float s = bh[c];
        #pragma unroll
        for (int p = 0; p < P_DIM; ++p)
            s += H3[b][p] * wh[p * C_DIM + c];
        out[(long)(b0 + b) * C_DIM + c] = s;
    }
}

extern "C" void kernel_launch(void* const* d_in, const int* in_sizes, int n_in,
                              void* d_out, int out_size, void* d_ws, size_t ws_size,
                              hipStream_t stream) {
    (void)in_sizes; (void)n_in; (void)out_size; (void)d_ws; (void)ws_size;
    const int*   x      = (const int*)  d_in[0];
    const float* emb    = (const float*)d_in[1];
    const float* omega1 = (const float*)d_in[2];
    const float* omega2 = (const float*)d_in[3];
    const float* w1     = (const float*)d_in[4];
    const float* b1     = (const float*)d_in[5];
    const float* w2     = (const float*)d_in[6];
    const float* b2     = (const float*)d_in[7];
    const float* wp     = (const float*)d_in[8];
    const float* bp     = (const float*)d_in[9];
    const float* wh     = (const float*)d_in[10];
    const float* bh     = (const float*)d_in[11];
    float*       out    = (float*)d_out;

    // B=128 -> 8 batch tiles of M=16 rows; 256 threads = 8 wave32 per tile
    donn_hopf_kernel<<<dim3(8), dim3(256), 0, stream>>>(
        x, emb, omega1, omega2, w1, b1, w2, b2, wp, bp, wh, bh, out);
}
